// GIN_5222680232278
// MI455X (gfx1250) — matrix operand order repaired and checked by reading, not compiled, not run on hardware
//
#include <hip/hip_runtime.h>
#include <hip/hip_bf16.h>

typedef __attribute__((ext_vector_type(16))) _Float16 v16h;
typedef __attribute__((ext_vector_type(8)))  _Float16 v8h;
typedef __attribute__((ext_vector_type(4)))  _Float16 v4h;
typedef __attribute__((ext_vector_type(8)))  float    v8f;

#define BN_EPS 1e-5f

// ---------------------------------------------------------------- utilities
__global__ void zero_kernel(float* __restrict__ p, long n) {
    long i = (long)blockIdx.x * blockDim.x + threadIdx.x;
    if (i < n) p[i] = 0.f;
}

// Pack one 64x64 fp32 weight matrix into WMMA B-fragment order (f16).
// Fragment (j = n-tile 0..3, c = k-chunk 0..1):
//   lane L, element e : n = j*16 + (L&15), k = c*32 + (e>>3)*16 + (L>>4)*8 + (e&7)
__global__ void pack64_kernel(const float* __restrict__ W, _Float16* __restrict__ Wf) {
    int idx = blockIdx.x * blockDim.x + threadIdx.x;   // 0..4095
    int e  = idx & 15;
    int L  = (idx >> 4) & 31;
    int jc = idx >> 9;                                  // 0..7
    int j = jc >> 1, c = jc & 1;
    int n = j * 16 + (L & 15);
    int k = c * 32 + (e >> 3) * 16 + (L >> 4) * 8 + (e & 7);
    Wf[idx] = (_Float16)W[k * 64 + n];
}

// ------------------------------------------------------------ edge scatter
// agg[dst] += h[src] ; 16 threads per edge, float4 each, f32 atomics (L2-resident)
__global__ void scatter_kernel(const float* __restrict__ hin, long rs,
                               const int* __restrict__ src, const int* __restrict__ dst,
                               float* __restrict__ agg, long E) {
    long idx = (long)blockIdx.x * blockDim.x + threadIdx.x;
    long e = idx >> 4;
    if (e >= E) return;
    int lane16 = (int)(idx & 15);
    long s = src[e], d = dst[e];
    const float4 v = *(const float4*)&hin[s * rs + lane16 * 4];
    float* ap = &agg[d * 64 + lane16 * 4];
    atomicAdd(ap + 0, v.x);
    atomicAdd(ap + 1, v.y);
    atomicAdd(ap + 2, v.z);
    atomicAdd(ap + 3, v.w);
}

// ------------------------------------------------- fused 2-layer GIN MLP (WMMA)
// One wave per 16-row node tile. z = h + agg -> f16 LDS tile ->
// [GEMM 64x64 -> bias -> BN -> ReLU] x2 ; second stage written to xjk slice.
// A layout (16-bit 16x32): lane L row m=L&15; elem e: k=(e>>3)*16+(L>>4)*8+(e&7)
// C/D layout (f32 16x16): lane L: n=L&15 (tile-local), reg r: m=(L>>4)*8+r
__launch_bounds__(32)
__global__ void gin_mlp_kernel(const float* __restrict__ hin, long rs_in,
                               const float* __restrict__ agg,
                               const _Float16* __restrict__ W1f,
                               const _Float16* __restrict__ W2f,
                               const float* __restrict__ b1, const float* __restrict__ g1,
                               const float* __restrict__ be1,
                               const float* __restrict__ b2, const float* __restrict__ g2,
                               const float* __restrict__ be2,
                               float* __restrict__ out, long rs_out) {
    __shared__ _Float16 tile[16][72];   // 16x64 f16, padded stride vs. bank conflicts
    const int  t    = threadIdx.x;
    const long base = (long)blockIdx.x * 16;
    const int  m    = t & 15;
    const int  hi   = t >> 4;
    const float inv = rsqrtf(1.f + BN_EPS);

    // Stage 0: z = h + agg  (fp32 -> f16 LDS tile)
    for (int q = t; q < 256; q += 32) {              // 16 rows x 16 float4
        int row = q >> 4, c4 = q & 15;
        const float4 hv = *(const float4*)&hin[(base + row) * rs_in + c4 * 4];
        const float4 av = *(const float4*)&agg[(base + row) * 64 + c4 * 4];
        v4h z;
        z[0] = (_Float16)(hv.x + av.x);
        z[1] = (_Float16)(hv.y + av.y);
        z[2] = (_Float16)(hv.z + av.z);
        z[3] = (_Float16)(hv.w + av.w);
        *(v4h*)&tile[row][c4 * 4] = z;
    }
    __syncthreads();

    for (int stage = 0; stage < 2; ++stage) {
        const _Float16* Wf = stage ? W2f : W1f;
        const float* bb = stage ? b2  : b1;
        const float* gg = stage ? g2  : g1;
        const float* be = stage ? be2 : be1;

        // A fragments for k-chunks 0 and 1 (two contiguous 8-half runs each)
        v8h l0 = *(const v8h*)&tile[m][hi * 8];
        v8h h0 = *(const v8h*)&tile[m][16 + hi * 8];
        v8h l1 = *(const v8h*)&tile[m][32 + hi * 8];
        v8h h1 = *(const v8h*)&tile[m][48 + hi * 8];
        v16h a0 = __builtin_shufflevector(l0, h0, 0,1,2,3,4,5,6,7,8,9,10,11,12,13,14,15);
        v16h a1 = __builtin_shufflevector(l1, h1, 0,1,2,3,4,5,6,7,8,9,10,11,12,13,14,15);

        v8f acc[4];
#pragma unroll
        for (int j = 0; j < 4; ++j) {
            v8f c = {};
            v16h b0 = *(const v16h*)&Wf[((j * 2 + 0) * 32 + t) * 16];
            v16h bv = *(const v16h*)&Wf[((j * 2 + 1) * 32 + t) * 16];
            c = __builtin_amdgcn_wmma_f32_16x16x32_f16(false, a0, false, b0,
                                                       (short)0, c, false, false);
            c = __builtin_amdgcn_wmma_f32_16x16x32_f16(false, a1, false, bv,
                                                       (short)0, c, false, false);
            acc[j] = c;
        }
        __syncthreads();   // all A-frag reads done before tile is overwritten

        if (stage == 0) {
#pragma unroll
            for (int j = 0; j < 4; ++j) {
                int n = j * 16 + m;
                float bvv = bb[n], gs = gg[n] * inv, bev = be[n];
#pragma unroll
                for (int r = 0; r < 8; ++r) {
                    float v = (acc[j][r] + bvv) * gs + bev;
                    tile[hi * 8 + r][n] = (_Float16)fmaxf(v, 0.f);
                }
            }
            __syncthreads();
        } else {
#pragma unroll
            for (int j = 0; j < 4; ++j) {
                int n = j * 16 + m;
                float bvv = bb[n], gs = gg[n] * inv, bev = be[n];
#pragma unroll
                for (int r = 0; r < 8; ++r) {
                    float v = (acc[j][r] + bvv) * gs + bev;
                    out[(base + hi * 8 + r) * rs_out + n] = fmaxf(v, 0.f);
                }
            }
        }
    }
}

// ------------------------------------------------------------ graph pooling
// batch is sorted -> per-thread running sums per dim, atomics only at graph edges
__global__ void pool_kernel(const float* __restrict__ xjk, const float* __restrict__ x,
                            const int* __restrict__ batch, float* __restrict__ pooled,
                            int N) {
    const int f  = threadIdx.x;            // 0..255 (feature)
    int i0 = blockIdx.x * 128;
    if (i0 >= N) return;
    int iend = min(i0 + 128, N);
    int g = batch[i0];
    float a0 = 0.f, a1 = 0.f, a2 = 0.f;
    for (int i = i0; i < iend; ++i) {
        int gi = batch[i];
        if (gi != g) {
            atomicAdd(&pooled[(0 * 128 + g) * 256 + f], a0);
            atomicAdd(&pooled[(1 * 128 + g) * 256 + f], a1);
            atomicAdd(&pooled[(2 * 128 + g) * 256 + f], a2);
            a0 = a1 = a2 = 0.f;
            g = gi;
        }
        float v = xjk[(long)i * 256 + f];
        a0 += v * x[(long)i * 64 + 0];
        a1 += v * x[(long)i * 64 + 1];
        a2 += v * x[(long)i * 64 + 2];
    }
    atomicAdd(&pooled[(0 * 128 + g) * 256 + f], a0);
    atomicAdd(&pooled[(1 * 128 + g) * 256 + f], a1);
    atomicAdd(&pooled[(2 * 128 + g) * 256 + f], a2);
}

// ----------------------------------------------------------------- readout
__global__ void head_kernel(const float* __restrict__ pooled,
                            const float* __restrict__ Wd, const float* __restrict__ bd,
                            const float* __restrict__ Wout, const float* __restrict__ bout,
                            float* __restrict__ out, int C) {
    __shared__ float rel[64];
    int g = blockIdx.x;     // 0..127
    int t = threadIdx.x;    // 0..63
    float acc = 0.f;
    for (int d = 0; d < 3; ++d) {
        const float* p = &pooled[(d * 128 + g) * 256];
        const float* W = &Wd[d * 256 * 64];
        float s = bd[d * 64 + t];
        for (int k = 0; k < 256; ++k) s += p[k] * W[k * 64 + t];
        acc += fmaxf(s, 0.f);
    }
    rel[t] = acc;
    __syncthreads();
    if (t < C) {
        float o = bout[t];
        for (int k = 0; k < 64; ++k) o += rel[k] * Wout[k * C + t];
        out[g * C + t] = o;
    }
}

// ------------------------------------------------------------------ launch
extern "C" void kernel_launch(void* const* d_in, const int* in_sizes, int n_in,
                              void* d_out, int out_size, void* d_ws, size_t ws_size,
                              hipStream_t stream) {
    const float* x    = (const float*)d_in[0];
    const int*   ei   = (const int*)d_in[1];
    const int*   bat  = (const int*)d_in[2];
    const float* W1   = (const float*)d_in[3];
    const float* b1   = (const float*)d_in[4];
    const float* g1   = (const float*)d_in[5];
    const float* be1  = (const float*)d_in[6];
    const float* W2   = (const float*)d_in[7];
    const float* b2   = (const float*)d_in[8];
    const float* g2   = (const float*)d_in[9];
    const float* be2  = (const float*)d_in[10];
    const float* Wd   = (const float*)d_in[11];
    const float* bd   = (const float*)d_in[12];
    const float* Wout = (const float*)d_in[13];
    const float* bout = (const float*)d_in[14];
    float* out = (float*)d_out;

    const long N    = in_sizes[0] / 64;          // 100000 (divisible by 16)
    const long E    = in_sizes[1] / 2;           // 3.2M
    const int  Lnum = in_sizes[3] / (64 * 64);   // 4
    const int  C    = in_sizes[14];              // 10

    char* ws = (char*)d_ws;
    float*    agg    = (float*)ws;    ws += (size_t)N * 64 * sizeof(float);
    float*    xjk    = (float*)ws;    ws += (size_t)N * 256 * sizeof(float);
    float*    pooled = (float*)ws;    ws += (size_t)3 * 128 * 256 * sizeof(float);
    _Float16* W1f    = (_Float16*)ws; ws += (size_t)Lnum * 4096 * sizeof(_Float16);
    _Float16* W2f    = (_Float16*)ws; ws += (size_t)Lnum * 4096 * sizeof(_Float16);

    for (int l = 0; l < Lnum; ++l) {
        pack64_kernel<<<16, 256, 0, stream>>>(W1 + l * 4096, W1f + l * 4096);
        pack64_kernel<<<16, 256, 0, stream>>>(W2 + l * 4096, W2f + l * 4096);
    }
    zero_kernel<<<(3 * 128 * 256 + 255) / 256, 256, 0, stream>>>(pooled, 3 * 128 * 256);

    const long nTiles = N / 16;
    for (int l = 0; l < Lnum; ++l) {
        const float* hin = (l == 0) ? x : (xjk + (long)(l - 1) * 64);
        long rs = (l == 0) ? 64 : 256;
        zero_kernel<<<(unsigned)((N * 64 + 255) / 256), 256, 0, stream>>>(agg, N * 64);
        long sthreads = E * 16;
        scatter_kernel<<<(unsigned)((sthreads + 255) / 256), 256, 0, stream>>>(
            hin, rs, ei, ei + E, agg, E);
        gin_mlp_kernel<<<(unsigned)nTiles, 32, 0, stream>>>(
            hin, rs, agg, W1f + l * 4096, W2f + l * 4096,
            b1 + l * 64, g1 + l * 64, be1 + l * 64,
            b2 + l * 64, g2 + l * 64, be2 + l * 64,
            xjk + (long)l * 64, 256);
    }
    pool_kernel<<<(unsigned)((N + 127) / 128), 256, 0, stream>>>(xjk, x, bat, pooled, (int)N);
    head_kernel<<<128, 64, 0, stream>>>(pooled, Wd, bd, Wout, bout, out, C);
}
